// MultiHeadedAttention_48301202210837
// MI455X (gfx1250) — matrix operand order repaired and checked by reading, not compiled, not run on hardware
//
#include <hip/hip_runtime.h>
#include <hip/hip_bf16.h>

// ---------------------------------------------------------------------------
// MultiHeadedAttention forward for MI455X (gfx1250, wave32, WMMA).
//
// ~104 GFLOP, working set << 192 MB L2 -> compute bound, so everything is
// routed through v_wmma_f32_16x16x32_f16. This revision removes per-tile
// f32->f16 conversions (one-time convert kernels for weights/activations)
// and widens the GEMM wave tile to 32x64 so each k-step issues 8 WMMAs from
// 8 contiguous per-lane loads.
// ---------------------------------------------------------------------------

typedef __attribute__((ext_vector_type(16))) _Float16 v16h;
typedef __attribute__((ext_vector_type(8)))  _Float16 v8h;
typedef __attribute__((ext_vector_type(8)))  float    v8f;
typedef __attribute__((ext_vector_type(4)))  float    v4f;

#define MHA_E  1024
#define MHA_H  16
#define MHA_HD 64
#define MHA_N  2
#define MHA_T  2048
#define MHA_S  2048
#define MHA_NEG (-1e9f)

enum { OUT_F16 = 0, OUT_F16_VT = 1, OUT_F32 = 2 };

// One-time f32 -> f16 conversion (8 elems/thread, vectorized).
__global__ __launch_bounds__(256) void cvt_f16_kernel(
    const float* __restrict__ in, _Float16* __restrict__ out, int n) {
  const int i = (blockIdx.x * 256 + threadIdx.x) * 8;
  if (i >= n) return;
  v4f x0 = *(const v4f*)(in + i);
  v4f x1 = *(const v4f*)(in + i + 4);
  v8h y;
#pragma unroll
  for (int j = 0; j < 4; ++j) {
    y[j]     = (_Float16)x0[j];
    y[4 + j] = (_Float16)x1[j];
  }
  *(v8h*)(out + i) = y;
}

// C(M,Nc) = A(M,Kd) @ W(Nc,Kd)^T + bias, all-f16 operands, f32 accum.
// Wave tile: 32(M) x 64(N) = 8 WMMA accumulators.
// Fragment layouts used here (contiguous k = 16h+s, not the ISA table k order):
//   A 16x32 f16 : lane l -> row l%16; K = k0+(l<16?0:16)+0..15
//                 => two contiguous v8h loads.
//   B 32x16 f16 : lane l -> N col l%16; K = k0+(l<16?0:16)+0..15
//                 => one contiguous v16h load of a W row slice.
//   C 16x16 f32 : vgpr r, lane l -> row r+(l<16?0:8), col l%16.
template <int OUT_MODE>
__global__ __launch_bounds__(128) void gemm_f16_wmma_kernel(
    const _Float16* __restrict__ A, const _Float16* __restrict__ W,
    const float* __restrict__ bias, void* __restrict__ Cv,
    int M, int Kd, int Nc, int rowsPerBatch) {
  const int wid  = (blockIdx.x * blockDim.x + threadIdx.x) >> 5;
  const int lane = threadIdx.x & 31;
  const int l16  = lane & 15;
  const int hs   = lane >> 4;

  const int ntiles = Nc >> 6;            // 64-wide N tiles
  const int nt = wid % ntiles;
  const int mt = wid / ntiles;
  if (mt * 32 >= M) return;              // wave-uniform: EXEC all-ones
  const int m0 = mt * 32, n0 = nt * 64;

  const long arow0 = (long)(m0 + l16) * Kd;
  const long arow1 = (long)(m0 + 16 + l16) * Kd;
  long wrow[4];
#pragma unroll
  for (int j = 0; j < 4; ++j) wrow[j] = (long)(n0 + j * 16 + l16) * Kd;

  v8f acc[2][4] = {};
  for (int k0 = 0; k0 < Kd; k0 += 32) {
    const int ka = k0 + hs * 16;         // lane's A chunk base
    const int kb = k0 + hs * 16;         // lane's B chunk base
    __builtin_prefetch((const void*)(A + arow0 + k0 + 512), 0, 1);

    v16h a0, a1;
    {
      v8h x0 = *(const v8h*)(A + arow0 + ka);
      v8h x1 = *(const v8h*)(A + arow0 + ka + 8);
      v8h x2 = *(const v8h*)(A + arow1 + ka);
      v8h x3 = *(const v8h*)(A + arow1 + ka + 8);
#pragma unroll
      for (int i = 0; i < 8; ++i) {
        a0[i] = x0[i]; a0[8 + i] = x1[i];
        a1[i] = x2[i]; a1[8 + i] = x3[i];
      }
    }
    v16h bf[4];
#pragma unroll
    for (int j = 0; j < 4; ++j) bf[j] = *(const v16h*)(W + wrow[j] + kb);

#pragma unroll
    for (int j = 0; j < 4; ++j) {
      acc[0][j] = __builtin_amdgcn_wmma_f32_16x16x32_f16(
          false, a0, false, bf[j], (short)0, acc[0][j], false, false);
      acc[1][j] = __builtin_amdgcn_wmma_f32_16x16x32_f16(
          false, a1, false, bf[j], (short)0, acc[1][j], false, false);
    }
  }

#pragma unroll
  for (int mi = 0; mi < 2; ++mi) {
#pragma unroll
    for (int j = 0; j < 4; ++j) {
      const int col = n0 + j * 16 + l16;
      const float bias_v = bias[col];
#pragma unroll
      for (int r = 0; r < 8; ++r) {
        const int row = m0 + mi * 16 + r + hs * 8;
        const float val = acc[mi][j][r] + bias_v;
        if (OUT_MODE == OUT_F16) {
          ((_Float16*)Cv)[(long)row * Nc + col] = (_Float16)val;
        } else if (OUT_MODE == OUT_F32) {
          ((float*)Cv)[(long)row * Nc + col] = val;
        } else {  // OUT_F16_VT: Vt[n][e][s] for contiguous P@V B-frag loads
          const int nb = row / rowsPerBatch;
          const int s  = row % rowsPerBatch;
          ((_Float16*)Cv)[((long)nb * Nc + col) * rowsPerBatch + s] =
              (_Float16)val;
        }
      }
    }
  }
}

// Flash attention: one wave per (n, h, 16 query rows); streams S in 32-col
// blocks up to the causal frontier; online softmax; P re-laid-out via LDS.
__global__ __launch_bounds__(128) void attn_wmma_kernel(
    const _Float16* __restrict__ Qh, const _Float16* __restrict__ Kh,
    const _Float16* __restrict__ Vt, const unsigned char* __restrict__ pad,
    _Float16* __restrict__ Yh) {
  constexpr int E = MHA_E, H = MHA_H, HD = MHA_HD, T = MHA_T, S = MHA_S;
  const int wslot = threadIdx.x >> 5;
  const int wid   = blockIdx.x * 4 + wslot;
  const int lane  = threadIdx.x & 31;
  const int l16   = lane & 15;
  const int hs    = lane >> 4;

  constexpr int TB = T / 16;
  const int tb = wid % TB;
  const int h  = (wid / TB) % H;
  const int n  = wid / (TB * H);
  const int t0 = tb * 16;

  __shared__ __align__(32) _Float16 ldsP[4][16 * 32];
  _Float16* P = &ldsP[wslot][0];

  // Q fragments for d=0..31 and d=32..63 (loaded once, reused every block)
  const _Float16* qbase = Qh + (long)(n * T + t0 + l16) * E + h * HD;
  v16h aq[2];
#pragma unroll
  for (int hd = 0; hd < 2; ++hd) {
    v8h x0 = *(const v8h*)(qbase + hd * 32 + hs * 16);
    v8h x1 = *(const v8h*)(qbase + hd * 32 + hs * 16 + 8);
#pragma unroll
    for (int i = 0; i < 8; ++i) { aq[hd][i] = x0[i]; aq[hd][8 + i] = x1[i]; }
  }

  float mrow[8], lrow[8];
#pragma unroll
  for (int r = 0; r < 8; ++r) { mrow[r] = -3.0e38f; lrow[r] = 0.0f; }
  v8f o[4] = {};

  for (int s0 = 0; s0 < t0 + 16; s0 += 32) {
    // scores (16 t x 32 s): two s-halves, K=64 folded as 2 WMMAs each
    v8f sc[2];
#pragma unroll
    for (int hsb = 0; hsb < 2; ++hsb) {
      v8f cacc = {};
      const _Float16* kb =
          Kh + (long)(n * S + s0 + hsb * 16 + l16) * E + h * HD + hs * 16;
#pragma unroll
      for (int hd = 0; hd < 2; ++hd) {
        v16h bk = *(const v16h*)(kb + hd * 32);
        cacc = __builtin_amdgcn_wmma_f32_16x16x32_f16(false, aq[hd], false, bk,
                                                      (short)0, cacc, false,
                                                      false);
      }
      sc[hsb] = cacc;
    }

    const int col0 = s0 + l16;
    const int col1 = s0 + 16 + l16;
    const bool pm0 = pad[n * S + col0] != 0;
    const bool pm1 = pad[n * S + col1] != 0;

#pragma unroll
    for (int r = 0; r < 8; ++r) {
      const int trow = t0 + r + hs * 8;
      float v0 = sc[0][r] * 0.125f;  // 1/sqrt(64)
      float v1 = sc[1][r] * 0.125f;
      if (pm0 || col0 > trow) v0 = MHA_NEG;  // causal == reference tril mask
      if (pm1 || col1 > trow) v1 = MHA_NEG;

      float mx = fmaxf(v0, v1);  // row reduce across the 16-lane half-group
      mx = fmaxf(mx, __shfl_xor(mx, 1, 32));
      mx = fmaxf(mx, __shfl_xor(mx, 2, 32));
      mx = fmaxf(mx, __shfl_xor(mx, 4, 32));
      mx = fmaxf(mx, __shfl_xor(mx, 8, 32));
      const float mnew  = fmaxf(mrow[r], mx);
      const float alpha = __expf(mrow[r] - mnew);
      const float e0 = __expf(v0 - mnew);
      const float e1 = __expf(v1 - mnew);
      float rs = e0 + e1;
      rs += __shfl_xor(rs, 1, 32);
      rs += __shfl_xor(rs, 2, 32);
      rs += __shfl_xor(rs, 4, 32);
      rs += __shfl_xor(rs, 8, 32);
      lrow[r] = lrow[r] * alpha + rs;
      mrow[r] = mnew;
#pragma unroll
      for (int dt = 0; dt < 4; ++dt) o[dt][r] *= alpha;

      // C-layout -> LDS (row-major 16x32 f16)
      P[(r + hs * 8) * 32 + l16]      = (_Float16)e0;
      P[(r + hs * 8) * 32 + 16 + l16] = (_Float16)e1;
    }

    // reload P in A-fragment layout (same wave: LDS in-order, DScnt waits)
    v16h pa;
    {
      const _Float16* pb = P + l16 * 32 + hs * 16;
      v8h y0 = *(const v8h*)(pb);
      v8h y1 = *(const v8h*)(pb + 8);
#pragma unroll
      for (int i = 0; i < 8; ++i) { pa[i] = y0[i]; pa[8 + i] = y1[i]; }
    }

    // O += P @ V ; Vt[n][e][s] makes the B-frag a contiguous 32B load
    const _Float16* vb =
        Vt + ((long)(n * H + h) * HD + l16) * S + s0 + hs * 16;
#pragma unroll
    for (int dt = 0; dt < 4; ++dt) {
      v16h bvv = *(const v16h*)(vb + (long)dt * 16 * S);
      o[dt] = __builtin_amdgcn_wmma_f32_16x16x32_f16(false, pa, false, bvv,
                                                     (short)0, o[dt], false,
                                                     false);
    }
  }

#pragma unroll
  for (int r = 0; r < 8; ++r) {
    const float inv = 1.0f / lrow[r];
    const long row = n * T + t0 + r + hs * 8;
#pragma unroll
    for (int dt = 0; dt < 4; ++dt) {
      Yh[row * E + h * HD + dt * 16 + l16] = (_Float16)(o[dt][r] * inv);
    }
  }
}

extern "C" void kernel_launch(void* const* d_in, const int* in_sizes, int n_in,
                              void* d_out, int out_size, void* d_ws,
                              size_t ws_size, hipStream_t stream) {
  (void)in_sizes; (void)n_in; (void)out_size; (void)ws_size;
  const float* query = (const float*)d_in[0];
  const float* key   = (const float*)d_in[1];
  const float* value = (const float*)d_in[2];
  // d_in[3] attn_mask == tril(ones) -> implemented analytically (causal)
  const unsigned char* pad = (const unsigned char*)d_in[4];
  const float* Wq = (const float*)d_in[5];
  const float* bq = (const float*)d_in[6];
  const float* Wk = (const float*)d_in[7];
  const float* bk = (const float*)d_in[8];
  const float* Wv = (const float*)d_in[9];
  const float* bv = (const float*)d_in[10];
  const float* Wp = (const float*)d_in[11];
  const float* bp = (const float*)d_in[12];
  float* out = (float*)d_out;

  const size_t actE = (size_t)MHA_N * MHA_T * MHA_E;  // 4M elems
  const size_t wE   = (size_t)MHA_E * MHA_E;          // 1M elems
  // Workspace layout (f16 elements), total 64 MiB:
  _Float16* Xq  = (_Float16*)d_ws;      // query f16
  _Float16* Xk  = Xq + actE;            // key f16
  _Float16* Xv  = Xk + actE;            // value f16
  _Float16* Wq6 = Xv + actE;            // weights f16
  _Float16* Wk6 = Wq6 + wE;
  _Float16* Wv6 = Wk6 + wE;
  _Float16* Wp6 = Wv6 + wE;
  _Float16* Qh  = Wp6 + wE;             // projected activations f16
  _Float16* Kh  = Qh + actE;
  _Float16* Vt  = Kh + actE;
  _Float16* Yh  = Vt + actE;

  // --- one-time f32 -> f16 conversions -------------------------------------
  const int actBlocks = (int)(actE / 8 / 256);
  const int wBlocks   = (int)(wE / 8 / 256);
  cvt_f16_kernel<<<actBlocks, 256, 0, stream>>>(query, Xq, (int)actE);
  cvt_f16_kernel<<<actBlocks, 256, 0, stream>>>(key,   Xk, (int)actE);
  cvt_f16_kernel<<<actBlocks, 256, 0, stream>>>(value, Xv, (int)actE);
  cvt_f16_kernel<<<wBlocks, 256, 0, stream>>>(Wq, Wq6, (int)wE);
  cvt_f16_kernel<<<wBlocks, 256, 0, stream>>>(Wk, Wk6, (int)wE);
  cvt_f16_kernel<<<wBlocks, 256, 0, stream>>>(Wv, Wv6, (int)wE);
  cvt_f16_kernel<<<wBlocks, 256, 0, stream>>>(Wp, Wp6, (int)wE);

  // --- projections ---------------------------------------------------------
  const int M = MHA_N * MHA_T;                       // 4096 rows
  const int gemmWaves  = (M / 32) * (MHA_E / 64);    // 2048
  const int gemmBlocks = (gemmWaves + 3) / 4;
  dim3 blk(128);

  gemm_f16_wmma_kernel<OUT_F16><<<gemmBlocks, blk, 0, stream>>>(
      Xq, Wq6, bq, Qh, M, MHA_E, MHA_E, MHA_T);
  gemm_f16_wmma_kernel<OUT_F16><<<gemmBlocks, blk, 0, stream>>>(
      Xk, Wk6, bk, Kh, M, MHA_E, MHA_E, MHA_S);
  gemm_f16_wmma_kernel<OUT_F16_VT><<<gemmBlocks, blk, 0, stream>>>(
      Xv, Wv6, bv, Vt, M, MHA_E, MHA_E, MHA_S);

  // --- attention -----------------------------------------------------------
  const int attnWaves  = MHA_N * MHA_H * (MHA_T / 16);  // 4096
  const int attnBlocks = (attnWaves + 3) / 4;
  attn_wmma_kernel<<<attnBlocks, blk, 0, stream>>>(Qh, Kh, Vt, pad, Yh);

  // --- output projection (f32 out) ----------------------------------------
  gemm_f16_wmma_kernel<OUT_F32><<<gemmBlocks, blk, 0, stream>>>(
      Yh, Wp6, bp, out, M, MHA_E, MHA_E, MHA_T);
}
